// MLGKALayer_22814866276999
// MI455X (gfx1250) — matrix operand order repaired and checked by reading, not compiled, not run on hardware
//
#include <hip/hip_runtime.h>

// ---------------------------------------------------------------------------
// MLGKA layer for MI455X (gfx1250): bf16 WMMA GEMMs (f32 accum) for all large
// matmuls, VALU kernels for RFF/attention-normalize/LayerNorm epilogues.
// ---------------------------------------------------------------------------

typedef __bf16 bf16_t;
typedef bf16_t bf16x16 __attribute__((ext_vector_type(16)));
typedef bf16_t bf16x8  __attribute__((ext_vector_type(8)));
typedef float  f32x8   __attribute__((ext_vector_type(8)));

#define TOK    16384   // B*S
#define DMODEL 2048
#define DFF    8192
#define NH     16
#define NKV    8
#define DH     128
#define SEQ    4096
#define NB     4
#define KDIM   64
#define FDIM   128     // 2*KDIM

// ---------------------------------------------------------------- fragments
__device__ __forceinline__ f32x8 f32x8_zero() {
  f32x8 z;
#pragma unroll
  for (int i = 0; i < 8; ++i) z[i] = 0.0f;
  return z;
}

// A fragment (16x32 bf16, MxK): lanes 0-15 / 16-31 both cover rows 0-15;
// per ISA: elems 0-7 -> K = 8*half + e, elems 8-15 -> K = 16 + 8*half + (e-8).
__device__ __forceinline__ bf16x16 frag_a(const bf16_t* __restrict__ tileBase,
                                          int ld, int lane) {
  const bf16_t* p = tileBase + (long)(lane & 15) * ld + ((lane >> 4) << 3);
  union { bf16x16 v; bf16x8 h[2]; } u;
  u.h[0] = *(const bf16x8*)p;
  u.h[1] = *(const bf16x8*)(p + 16);
  return u.v;
}

// B fragment (32x16 bf16, KxN) sourced from Bt stored [N,K] row-major:
// lane%16 selects column n; elems 0-7 -> K = 8*half + e, elems 8-15 -> K = 16 + 8*half + (e-8).
__device__ __forceinline__ bf16x16 frag_b(const bf16_t* __restrict__ Bt, int ld,
                                          int n0, int N, int kk, int lane) {
  int row = n0 + (lane & 15);
  row = row < N ? row : N - 1;               // clamp for N<tile edges
  const bf16_t* p = Bt + (long)row * ld + kk + ((lane >> 4) << 3);
  union { bf16x16 v; bf16x8 h[2]; } u;
  u.h[0] = *(const bf16x8*)p;
  u.h[1] = *(const bf16x8*)(p + 16);
  return u.v;
}

__device__ __forceinline__ f32x8 wmma_bf16(bf16x16 a, bf16x16 b, f32x8 c) {
  return __builtin_amdgcn_wmma_f32_16x16x32_bf16(
      /*neg_a=*/false, a, /*neg_b=*/false, b,
      /*c_mod=*/(short)0, c, /*reuse_a=*/false, /*reuse_b=*/false);
}

// ------------------------------------------------------ generic batched GEMM
// C[M,N] = A[M,K] (bf16, row major) x Bt[N,K]^T (bf16). 256 thr = 8 waves,
// block tile 128x128, wave tile 32x64 (2x4 WMMA tiles), K-step 32.
__global__ __launch_bounds__(256)
void mlgka_gemm_bf16(const bf16_t* __restrict__ A, const bf16_t* __restrict__ Bt,
                     float* __restrict__ Cf, bf16_t* __restrict__ Cb,
                     int M, int N, int K, int ldc,
                     long aBatch, long bBatch, long cBatch) {
  int batch = blockIdx.z;
  A  += (long)batch * aBatch;
  Bt += (long)batch * bBatch;
  long cbase = (long)batch * cBatch;

  int wave = threadIdx.x >> 5, lane = threadIdx.x & 31;
  int wm = blockIdx.x * 128 + (wave >> 1) * 32;   // wave row base
  int wn = blockIdx.y * 128 + (wave & 1) * 64;    // wave col base

  f32x8 acc[2][4];
#pragma unroll
  for (int i = 0; i < 2; ++i)
#pragma unroll
    for (int j = 0; j < 4; ++j) acc[i][j] = f32x8_zero();

  for (int kk = 0; kk < K; kk += 32) {
    bf16x16 af[2], bfv[4];
#pragma unroll
    for (int i = 0; i < 2; ++i)
      af[i] = frag_a(A + (long)(wm + i * 16) * K + kk, K, lane);
#pragma unroll
    for (int j = 0; j < 4; ++j)
      bfv[j] = frag_b(Bt, K, wn + j * 16, N, kk, lane);

    if (kk + 32 < K) {   // pull next K-step into cache (global_prefetch_b8)
      __builtin_prefetch(A + (long)(wm + (lane & 15)) * K + kk + 32, 0, 1);
      int pr = wn + (lane & 15); pr = pr < N ? pr : N - 1;
      __builtin_prefetch(Bt + (long)pr * K + kk + 32, 0, 1);
    }
#pragma unroll
    for (int i = 0; i < 2; ++i)
#pragma unroll
      for (int j = 0; j < 4; ++j)
        acc[i][j] = wmma_bf16(af[i], bfv[j], acc[i][j]);
  }

  int half = lane >> 4, nl = lane & 15;
#pragma unroll
  for (int i = 0; i < 2; ++i)
#pragma unroll
    for (int j = 0; j < 4; ++j) {
      int col = wn + j * 16 + nl;
      if (col < N) {
#pragma unroll
        for (int jv = 0; jv < 8; ++jv) {
          int row = wm + i * 16 + jv + 8 * half;
          long ci = cbase + (long)row * ldc + col;
          float val = acc[i][j][jv];
          if (Cf) Cf[ci] = val;
          if (Cb) Cb[ci] = (bf16_t)val;
        }
      }
    }
}

// ---------------------------------------------- fused SwiGLU GEMM (gate & up)
// out[M,DFF] = (A x Wg^T) * silu(A x Wu^T), bf16 out. Block tile 128x64,
// wave tile 32x32 per matrix (2x2 WMMA tiles each -> 8 wmma / K-step).
__global__ __launch_bounds__(256)
void mlgka_gemm_swiglu(const bf16_t* __restrict__ A, const bf16_t* __restrict__ Bg,
                       const bf16_t* __restrict__ Bu, bf16_t* __restrict__ out,
                       int K, int NFF) {
  int wave = threadIdx.x >> 5, lane = threadIdx.x & 31;
  int wm = blockIdx.x * 128 + (wave >> 1) * 32;
  int wn = blockIdx.y * 64 + (wave & 1) * 32;

  f32x8 accg[2][2], accu[2][2];
#pragma unroll
  for (int i = 0; i < 2; ++i)
#pragma unroll
    for (int j = 0; j < 2; ++j) { accg[i][j] = f32x8_zero(); accu[i][j] = f32x8_zero(); }

  for (int kk = 0; kk < K; kk += 32) {
    bf16x16 af[2], bg[2], bu[2];
#pragma unroll
    for (int i = 0; i < 2; ++i)
      af[i] = frag_a(A + (long)(wm + i * 16) * K + kk, K, lane);
#pragma unroll
    for (int j = 0; j < 2; ++j) {
      bg[j] = frag_b(Bg, K, wn + j * 16, NFF, kk, lane);
      bu[j] = frag_b(Bu, K, wn + j * 16, NFF, kk, lane);
    }
    if (kk + 32 < K) {
      __builtin_prefetch(A + (long)(wm + (lane & 15)) * K + kk + 32, 0, 1);
      __builtin_prefetch(Bg + (long)(wn + (lane & 15)) * K + kk + 32, 0, 1);
      __builtin_prefetch(Bu + (long)(wn + (lane & 15)) * K + kk + 32, 0, 1);
    }
#pragma unroll
    for (int i = 0; i < 2; ++i)
#pragma unroll
      for (int j = 0; j < 2; ++j) {
        accg[i][j] = wmma_bf16(af[i], bg[j], accg[i][j]);
        accu[i][j] = wmma_bf16(af[i], bu[j], accu[i][j]);
      }
  }

  int half = lane >> 4, nl = lane & 15;
#pragma unroll
  for (int i = 0; i < 2; ++i)
#pragma unroll
    for (int j = 0; j < 2; ++j) {
#pragma unroll
      for (int jv = 0; jv < 8; ++jv) {
        int row = wm + i * 16 + jv + 8 * half;
        int col = wn + j * 16 + nl;
        float g = accg[i][j][jv], u = accu[i][j][jv];
        float silu = u / (1.0f + __expf(-u));
        out[(long)row * NFF + col] = (bf16_t)(g * silu);
      }
    }
}

// ------------------------------------------------------------- small kernels
__global__ __launch_bounds__(256)
void mlgka_f32_to_bf16(const float* __restrict__ src, bf16_t* __restrict__ dst, long n) {
  long i = (long)blockIdx.x * 256 + threadIdx.x;
  long stride = (long)gridDim.x * 256;
  for (; i < n; i += stride) dst[i] = (bf16_t)src[i];
}

// dst[cols,rows] (bf16) = transpose(src[rows,cols] f32)
__global__ __launch_bounds__(256)
void mlgka_transpose_to_bf16(const float* __restrict__ src, bf16_t* __restrict__ dst,
                             int rows, int cols) {
  __shared__ float tile[32][33];
  int bx = blockIdx.x * 32;        // col base
  int by = blockIdx.y * 32;        // row base
  int tx = threadIdx.x & 31, ty = threadIdx.x >> 5;   // 32x8
  for (int i = 0; i < 32; i += 8) {
    int r = by + ty + i, c = bx + tx;
    if (r < rows && c < cols) tile[ty + i][tx] = src[(long)r * cols + c];
  }
  __syncthreads();
  for (int i = 0; i < 32; i += 8) {
    int r = bx + ty + i, c = by + tx;
    if (r < cols && c < rows) dst[(long)r * rows + c] = (bf16_t)tile[tx][ty + i];
  }
}

// Laplacian-kernel random Fourier features: one block per token.
// src[T, nsrcH*128] f32 -> dst[B,H,S,128] bf16 (cos|sin)*1/sqrt(64).
__global__ __launch_bounds__(256)
void mlgka_phi(const float* __restrict__ src, const float* __restrict__ omega,
               bf16_t* __restrict__ dst, int srcRowLen, int rep) {
  __shared__ float om[DH * KDIM];     // 32 KB
  __shared__ float row[DMODEL];       // 8 KB
  int t = blockIdx.x, tid = threadIdx.x;
  for (int i = tid; i < DH * KDIM; i += 256) om[i] = omega[i];
  for (int i = tid; i < srcRowLen; i += 256) row[i] = src[(long)t * srcRowLen + i];
  __syncthreads();
  int b = t / SEQ, s = t % SEQ;
  for (int e = tid; e < NH * KDIM; e += 256) {
    int h = e >> 6, kd = e & 63;
    const float* qr = &row[(h / rep) * DH];
    float p = 0.0f;
#pragma unroll 8
    for (int d = 0; d < DH; ++d) p += qr[d] * om[d * KDIM + kd];
    long base = (((long)(b * NH + h) * SEQ) + s) * FDIM;
    dst[base + kd]         = (bf16_t)(__cosf(p) * 0.125f);
    dst[base + KDIM + kd]  = (bf16_t)(__sinf(p) * 0.125f);
  }
}

// Per (b,h): kvT[d][f] = sum_s phi_k[s,f]*v[s,d] (bf16 out), z[f] = sum_s phi_k.
__global__ __launch_bounds__(256)
void mlgka_kvz(const bf16_t* __restrict__ phi_k, const float* __restrict__ v,
               bf16_t* __restrict__ kvT, float* __restrict__ z) {
  __shared__ float vv[DH];
  __shared__ float pk[FDIM];
  int bh = blockIdx.x;
  int b = bh / NH, h = bh % NH, kvh = h >> 1;   // GQA rep = 2
  int tid = threadIdx.x;
  int f = tid >> 1, d0 = (tid & 1) * 64;
  float acc[64];
#pragma unroll
  for (int i = 0; i < 64; ++i) acc[i] = 0.0f;
  float zacc = 0.0f;
  for (int s = 0; s < SEQ; ++s) {
    if (tid < DH)      vv[tid] = v[(long)(b * SEQ + s) * (NKV * DH) + kvh * DH + tid];
    else               pk[tid - DH] =
        (float)phi_k[(((long)(b * NH + h) * SEQ) + s) * FDIM + (tid - DH)];
    __syncthreads();
    float p = pk[f];
#pragma unroll 8
    for (int d = 0; d < 64; ++d) acc[d] += p * vv[d0 + d];
    if ((tid & 1) == 0) zacc += p;
    __syncthreads();
  }
  long kb = (long)bh * DH * FDIM;
#pragma unroll 8
  for (int d = 0; d < 64; ++d) kvT[kb + (long)(d0 + d) * FDIM + f] = (bf16_t)acc[d];
  if ((tid & 1) == 0) z[(long)bh * FDIM + f] = zacc;
}

// attn[t, h*128+d] = num[b,h,s,d] / (phi_q . z + 1e-6), bf16 out (one thr per (t,h)).
__global__ __launch_bounds__(256)
void mlgka_combine(const bf16_t* __restrict__ phi_q, const float* __restrict__ num,
                   const float* __restrict__ z, bf16_t* __restrict__ attn) {
  long idx = (long)blockIdx.x * 256 + threadIdx.x;   // T*NH
  int h = (int)(idx & 15);
  long t = idx >> 4;
  int b = (int)(t / SEQ), s = (int)(t % SEQ);
  long base = (((long)(b * NH + h) * SEQ) + s) * FDIM;
  const float* zz = &z[(long)(b * NH + h) * FDIM];
  float den = 0.0f;
#pragma unroll 8
  for (int f = 0; f < FDIM; ++f) den += (float)phi_q[base + f] * zz[f];
  float inv = 1.0f / (den + 1e-6f);
  bf16_t* out = &attn[t * DMODEL + h * DH];
#pragma unroll 8
  for (int d = 0; d < DH; ++d) out[d] = (bf16_t)(num[base + d] * inv);
}

// y = LN(x + r)*g + b; writes f32 (residual) and/or bf16 (next GEMM A).
__global__ __launch_bounds__(256)
void mlgka_add_ln(const float* __restrict__ x, const float* __restrict__ r,
                  const float* __restrict__ g, const float* __restrict__ bta,
                  float* __restrict__ outF, bf16_t* __restrict__ outB) {
  __shared__ float s1[256], s2[256];
  int t = blockIdx.x, tid = threadIdx.x;
  float v[8];
  float sum = 0.0f, sq = 0.0f;
#pragma unroll
  for (int i = 0; i < 8; ++i) {
    int e = tid + i * 256;
    float val = x[(long)t * DMODEL + e] + r[(long)t * DMODEL + e];
    v[i] = val; sum += val; sq += val * val;
  }
  s1[tid] = sum; s2[tid] = sq;
  __syncthreads();
  for (int st = 128; st > 0; st >>= 1) {
    if (tid < st) { s1[tid] += s1[tid + st]; s2[tid] += s2[tid + st]; }
    __syncthreads();
  }
  float mu = s1[0] * (1.0f / DMODEL);
  float var = s2[0] * (1.0f / DMODEL) - mu * mu;
  float rstd = rsqrtf(var + 1e-5f);
#pragma unroll
  for (int i = 0; i < 8; ++i) {
    int e = tid + i * 256;
    float y = (v[i] - mu) * rstd * g[e] + bta[e];
    if (outF) outF[(long)t * DMODEL + e] = y;
    if (outB) outB[(long)t * DMODEL + e] = (bf16_t)y;
  }
}

// ---------------------------------------------------------------- launcher
extern "C" void kernel_launch(void* const* d_in, const int* in_sizes, int n_in,
                              void* d_out, int out_size, void* d_ws, size_t ws_size,
                              hipStream_t stream) {
  (void)in_sizes; (void)n_in; (void)out_size; (void)ws_size;
  const float* x      = (const float*)d_in[0];
  const float* W_dq   = (const float*)d_in[1];
  const float* W_uq   = (const float*)d_in[2];
  const float* W_dkv  = (const float*)d_in[3];
  const float* W_uk   = (const float*)d_in[4];
  const float* W_uv   = (const float*)d_in[5];
  const float* omega  = (const float*)d_in[6];
  const float* W_o    = (const float*)d_in[7];
  const float* ln1_g  = (const float*)d_in[8];
  const float* ln1_b  = (const float*)d_in[9];
  const float* gate_W = (const float*)d_in[10];
  const float* up_W   = (const float*)d_in[11];
  const float* down_W = (const float*)d_in[12];
  const float* ln2_g  = (const float*)d_in[13];
  const float* ln2_b  = (const float*)d_in[14];
  float* out = (float*)d_out;

  // ---- workspace layout (bump allocator, explicit region reuse, ~855 MB) ----
  char* ws = (char*)d_ws;
  size_t off = 0;
  auto take = [&](size_t bytes) -> char* {
    char* p = ws + off; off += (bytes + 255) & ~(size_t)255; return p;
  };
  bf16_t* wdqT  = (bf16_t*)take((size_t)64 * 2048 * 2);
  bf16_t* wuqT  = (bf16_t*)take((size_t)2048 * 64 * 2);
  bf16_t* wdkvT = (bf16_t*)take((size_t)64 * 2048 * 2);
  bf16_t* wukT  = (bf16_t*)take((size_t)1024 * 64 * 2);
  bf16_t* wuvT  = (bf16_t*)take((size_t)1024 * 64 * 2);
  bf16_t* woT   = (bf16_t*)take((size_t)2048 * 2048 * 2);
  bf16_t* gateT = (bf16_t*)take((size_t)8192 * 2048 * 2);
  bf16_t* upT   = (bf16_t*)take((size_t)8192 * 2048 * 2);
  bf16_t* downT = (bf16_t*)take((size_t)2048 * 8192 * 2);
  char*   rXB   = take((size_t)TOK * DMODEL * 2);          // xb, later attn bf16
  bf16_t* cqb   = (bf16_t*)take((size_t)TOK * 64 * 2);
  bf16_t* ckvb  = (bf16_t*)take((size_t)TOK * 64 * 2);
  char*   rQKV  = take((size_t)TOK * DMODEL * 4 * 2);      // q|k|v f32, later gated bf16
  char*   rPHI  = take((size_t)TOK * NH * FDIM * 2 * 2 / 2); // phi_q|phi_k bf16 (67MB), later attnout f32
  bf16_t* kvT   = (bf16_t*)take((size_t)NB * NH * DH * FDIM * 2);
  float*  zbuf  = (float*)take((size_t)NB * NH * FDIM * 4);
  char*   rNUM  = take((size_t)TOK * NH * FDIM * 4 / 2 * 2); // num f32 (134MB), later ffn f32
  float*  hbuf  = (float*)take((size_t)TOK * DMODEL * 4);
  bf16_t* hb    = (bf16_t*)take((size_t)TOK * DMODEL * 2);

  bf16_t* xb      = (bf16_t*)rXB;
  bf16_t* attn    = (bf16_t*)rXB;                              // reuse after GEMM1/3
  float*  q       = (float*)rQKV;
  float*  k       = (float*)(rQKV + (size_t)TOK * DMODEL * 4);
  float*  v       = (float*)(rQKV + (size_t)TOK * DMODEL * 4 + (size_t)TOK * 1024 * 4);
  bf16_t* gated   = (bf16_t*)rQKV;                             // reuse after kvz
  bf16_t* phi_q   = (bf16_t*)rPHI;
  bf16_t* phi_k   = (bf16_t*)(rPHI + (size_t)TOK * NH * FDIM * 2 / 2);
  float*  attnout = (float*)rPHI;                              // reuse after combine
  float*  num     = (float*)rNUM;
  float*  ffn     = (float*)rNUM;                              // reuse after combine

  auto cdiv = [](int a, int b) { return (a + b - 1) / b; };

  // ---- stage 0: conversions / weight transposes (bf16 [N,K] for WMMA B) ----
  mlgka_f32_to_bf16<<<4096, 256, 0, stream>>>(x, xb, (long)TOK * DMODEL);
  mlgka_transpose_to_bf16<<<dim3(cdiv(64,32),  cdiv(2048,32)), 256, 0, stream>>>(W_dq, wdqT, 2048, 64);
  mlgka_transpose_to_bf16<<<dim3(cdiv(2048,32),cdiv(64,32)),   256, 0, stream>>>(W_uq, wuqT, 64, 2048);
  mlgka_transpose_to_bf16<<<dim3(cdiv(64,32),  cdiv(2048,32)), 256, 0, stream>>>(W_dkv, wdkvT, 2048, 64);
  mlgka_transpose_to_bf16<<<dim3(cdiv(1024,32),cdiv(64,32)),   256, 0, stream>>>(W_uk, wukT, 64, 1024);
  mlgka_transpose_to_bf16<<<dim3(cdiv(1024,32),cdiv(64,32)),   256, 0, stream>>>(W_uv, wuvT, 64, 1024);
  mlgka_transpose_to_bf16<<<dim3(cdiv(2048,32),cdiv(2048,32)), 256, 0, stream>>>(W_o, woT, 2048, 2048);
  mlgka_transpose_to_bf16<<<dim3(cdiv(8192,32),cdiv(2048,32)), 256, 0, stream>>>(gate_W, gateT, 2048, 8192);
  mlgka_transpose_to_bf16<<<dim3(cdiv(8192,32),cdiv(2048,32)), 256, 0, stream>>>(up_W, upT, 2048, 8192);
  mlgka_transpose_to_bf16<<<dim3(cdiv(2048,32),cdiv(8192,32)), 256, 0, stream>>>(down_W, downT, 8192, 2048);

  // ---- stage 1: low-rank projections ----
  mlgka_gemm_bf16<<<dim3(TOK/128, 1, 1), 256, 0, stream>>>(xb, wdqT, nullptr, cqb,
      TOK, 64, 2048, 64, 0, 0, 0);
  mlgka_gemm_bf16<<<dim3(TOK/128, 1, 1), 256, 0, stream>>>(xb, wdkvT, nullptr, ckvb,
      TOK, 64, 2048, 64, 0, 0, 0);
  mlgka_gemm_bf16<<<dim3(TOK/128, 16, 1), 256, 0, stream>>>(cqb, wuqT, q, nullptr,
      TOK, 2048, 64, 2048, 0, 0, 0);
  mlgka_gemm_bf16<<<dim3(TOK/128, 8, 1), 256, 0, stream>>>(ckvb, wukT, k, nullptr,
      TOK, 1024, 64, 1024, 0, 0, 0);
  mlgka_gemm_bf16<<<dim3(TOK/128, 8, 1), 256, 0, stream>>>(ckvb, wuvT, v, nullptr,
      TOK, 1024, 64, 1024, 0, 0, 0);

  // ---- stage 2: RFF features, kv/z, num, normalize ----
  mlgka_phi<<<TOK, 256, 0, stream>>>(q, omega, phi_q, NH * DH, 1);
  mlgka_phi<<<TOK, 256, 0, stream>>>(k, omega, phi_k, NKV * DH, 2);
  mlgka_kvz<<<NB * NH, 256, 0, stream>>>(phi_k, v, kvT, zbuf);
  // num[b,h,s,:] = phi_q @ kv : batched WMMA GEMM over 64 (b,h) pairs
  mlgka_gemm_bf16<<<dim3(SEQ/128, 1, NB * NH), 256, 0, stream>>>(phi_q, kvT, num, nullptr,
      SEQ, DH, FDIM, DH, (long)SEQ * FDIM, (long)DH * FDIM, (long)SEQ * DH);
  mlgka_combine<<<(TOK * NH) / 256, 256, 0, stream>>>(phi_q, num, zbuf, attn);

  // ---- stage 3: output projection + residual + LN1 ----
  mlgka_gemm_bf16<<<dim3(TOK/128, 16, 1), 256, 0, stream>>>(attn, woT, attnout, nullptr,
      TOK, 2048, 2048, 2048, 0, 0, 0);
  mlgka_add_ln<<<TOK, 256, 0, stream>>>(x, attnout, ln1_g, ln1_b, hbuf, hb);

  // ---- stage 4: fused SwiGLU FFN + residual + LN2 ----
  mlgka_gemm_swiglu<<<dim3(TOK/128, DFF/64, 1), 256, 0, stream>>>(hb, gateT, upT, gated,
      DMODEL, DFF);
  mlgka_gemm_bf16<<<dim3(TOK/128, 16, 1), 256, 0, stream>>>(gated, downT, ffn, nullptr,
      TOK, 2048, 8192, 2048, 0, 0, 0);
  mlgka_add_ln<<<TOK, 256, 0, stream>>>(hbuf, ffn, ln2_g, ln2_b, out, nullptr);
}
